// MultiHeadAttention_6176162972316
// MI455X (gfx1250) — matrix operand order repaired and checked by reading, not compiled, not run on hardware
//
#include <hip/hip_runtime.h>
#include <hip/hip_bf16.h>

// ---------------------------------------------------------------------------
// MI455X (gfx1250) Multi-Head Attention, bf16 WMMA path.
//   - all GEMMs via v_wmma_f32_16x16x32_bf16 (f32 accumulate)
//   - projection GEMMs: 128x64 block tile, TDM (tensor_load_to_lds) staged,
//     double-buffered LDS, 2x2 WMMA register blocking per wave
//   - V stored transposed per head so the PV GEMM B-fragments are contiguous
// ---------------------------------------------------------------------------

typedef __attribute__((ext_vector_type(16))) __bf16 v16bf;
typedef __attribute__((ext_vector_type(8)))  float  v8f;

#if __has_builtin(__builtin_amdgcn_tensor_load_to_lds) && \
    __has_builtin(__builtin_amdgcn_s_wait_tensorcnt)
#define USE_TDM 1
#endif

__device__ __forceinline__ unsigned short f2bf(float f) {
    unsigned int u = __float_as_uint(f);
    u += 0x7fffu + ((u >> 16) & 1u);          // round-to-nearest-even
    return (unsigned short)(u >> 16);
}

// A-fragment (16x32 bf16, MxK): lane m=lane&15, half=lane>>4.
// element e: K = half*16 + e  -> 16 contiguous K elements (matches B).
__device__ __forceinline__ v16bf load_frag_a(const unsigned short* p, int half) {
    union { v16bf v; uint4 q[2]; } u;
    u.q[0] = *(const uint4*)(p + half * 16);
    u.q[1] = *(const uint4*)(p + half * 16 + 8);
    return u.v;
}

// B-fragment (32x16 bf16, KxN): lane n=lane&15, kset=(lane>>4)*16.
// 16 contiguous K elements per lane (caller bakes kset into pointer).
__device__ __forceinline__ v16bf load_frag_b(const unsigned short* p) {
    union { v16bf v; uint4 q[2]; } u;
    u.q[0] = *(const uint4*)(p);
    u.q[1] = *(const uint4*)(p + 8);
    return u.v;
}

__device__ __forceinline__ v8f wmma_bf16(v16bf a, v16bf b, v8f c) {
    return __builtin_amdgcn_wmma_f32_16x16x32_bf16(
        /*neg_a=*/false, a, /*neg_b=*/false, b,
        /*c_mod=*/(short)0, c, /*reuse_a=*/false, /*reuse_b=*/false);
}

#ifdef USE_TDM
typedef __attribute__((ext_vector_type(4))) unsigned int v4u_;
typedef __attribute__((ext_vector_type(8))) int          v8i_;
typedef __attribute__((ext_vector_type(4))) int          v4i_;

// 2D tile DMA: tile_rows x 32 bf16, source row stride = row_stride elements.
// D# built per CDNA5 ISA 8.3/8.4 (group0: count|lds|global|type,
// group1: data_size, tensor dims, tile dims, dim0 stride).
// This toolchain exposes the 6-arg builtin:
//   (uint32x4 g0, int32x8 g1, int32x4 g2, int32x4 g3, int32x8 pad, i32 cpol)
__device__ __forceinline__ void tdm_load_tile(const unsigned short* gsrc,
                                              unsigned lds_byte_off,
                                              int tile_rows, int row_stride) {
    unsigned long long ga = (unsigned long long)gsrc;
    v4u_ g0 = { 1u,                                    // count=1, user desc
                lds_byte_off,                          // lds_addr
                (unsigned)(ga & 0xffffffffu),          // global_addr[31:0]
                (unsigned)((ga >> 32) & 0x01ffffffu)   // global_addr[56:32]
                    | 0x80000000u };                   // type=2 (bits 127:126)
    v8i_ g1 = {
        (int)0x00010000u,                              // data_size=1 -> 2 bytes
        (int)((1024u & 0xffffu) << 16),                // tensor_dim0 lo16 << 16
        (int)(((1024u >> 16) & 0xffffu) |
              ((4096u & 0xffffu) << 16)),              // dim0 hi | dim1 lo << 16
        (int)(((4096u >> 16) & 0xffffu) | (32u << 16)),// dim1 hi | tile_dim0=32
        (int)(unsigned)tile_rows,                      // tile_dim1 (tile_dim2=0)
        (int)(unsigned)row_stride,                     // tensor_dim0_stride lo32
        0, 0 };
    v4i_ z4 = { 0, 0, 0, 0 };
    v8i_ z8 = { 0, 0, 0, 0, 0, 0, 0, 0 };
    __builtin_amdgcn_tensor_load_to_lds(g0, g1, z4, z4, z8, 0);
}
#endif

// ---------------------------------------------------------------------------
// f32 -> bf16 conversion
// ---------------------------------------------------------------------------
__global__ __launch_bounds__(256) void cvt_bf16_kernel(
    const float* __restrict__ in, unsigned short* __restrict__ out, int n) {
    int i = blockIdx.x * 256 + threadIdx.x;
    if (i < n) out[i] = f2bf(in[i]);
}

// ---------------------------------------------------------------------------
// C = (A[M,K] @ B[N,K]^T + bias[N]) * alpha,   M=4096, N=1024, K=1024
// MODE 0: f32 [M,N]   MODE 1: bf16 [M,N]   MODE 2: bf16 per-head transposed
// Block tile 128x64, 8 waves (4x2), wave tile 32x32 (2x2 WMMA accumulators).
// A/B tiles staged in double-buffered LDS via TDM (fallback: coop copy).
// ---------------------------------------------------------------------------
template <int MODE>
__global__ __launch_bounds__(256) void gemm_bf16_tn_kernel(
    const unsigned short* __restrict__ A,
    const unsigned short* __restrict__ Bm,
    const float* __restrict__ bias,
    void* __restrict__ Cout,
    float alpha) {
    constexpr int N = 1024, K = 1024;
    constexpr int BM = 128, BN = 64, BK = 32;

    // LDS layout is static -> known byte offsets for the TDM descriptors.
    __shared__ __align__(16) struct {
        unsigned short A[2][BM * BK];   // 8 KB per buffer, offset buf*8192
        unsigned short B[2][BN * BK];   // 4 KB per buffer, offset 16384+buf*4096
    } sm;

    const int lane = threadIdx.x & 31;
    const int wave = threadIdx.x >> 5;
    const int half = lane >> 4;
    const int ln   = lane & 15;
    const int wm   = wave >> 1;        // 0..3
    const int wn   = wave & 1;         // 0..1

    const int m0 = blockIdx.x * BM;
    const int n0 = blockIdx.y * BN;
    const unsigned short* Ab = A  + (size_t)m0 * K;
    const unsigned short* Bb = Bm + (size_t)n0 * K;

    v8f acc00 = {}, acc01 = {}, acc10 = {}, acc11 = {};

#ifdef USE_TDM
    if (wave == 0) {
        tdm_load_tile(Ab, 0u, BM, K);
        tdm_load_tile(Bb, 16384u, BN, K);
    }
    int cur = 0;
    for (int kb = 0; kb < K; kb += BK) {
        if (wave == 0) __builtin_amdgcn_s_wait_tensorcnt(0);
        __syncthreads();                       // tile `cur` visible to all
        if (wave == 0 && kb + BK < K) {        // overlap next DMA with compute
            const int nb = cur ^ 1;
            tdm_load_tile(Ab + kb + BK, (unsigned)(nb * 8192), BM, K);
            tdm_load_tile(Bb + kb + BK, (unsigned)(16384 + nb * 4096), BN, K);
        }
        {
            const unsigned short* sa = sm.A[cur];
            const unsigned short* sb = sm.B[cur];
            v16bf a0 = load_frag_a(sa + (wm * 32 + ln) * BK, half);
            v16bf a1 = load_frag_a(sa + (wm * 32 + 16 + ln) * BK, half);
            v16bf b0 = load_frag_b(sb + (wn * 32 + ln) * BK + half * 16);
            v16bf b1 = load_frag_b(sb + (wn * 32 + 16 + ln) * BK + half * 16);
            acc00 = wmma_bf16(a0, b0, acc00);
            acc01 = wmma_bf16(a0, b1, acc01);
            acc10 = wmma_bf16(a1, b0, acc10);
            acc11 = wmma_bf16(a1, b1, acc11);
        }
        __syncthreads();                       // done reading `cur`
        cur ^= 1;
    }
#else
    for (int kb = 0; kb < K; kb += BK) {
        // cooperative tile copy: A = 512 uint4, B = 256 uint4, 256 threads
        const int t = threadIdx.x;
        {
            int u0 = t, u1 = t + 256;          // A chunks (row = u>>2, c = u&3)
            *(uint4*)&sm.A[0][(u0 >> 2) * BK + (u0 & 3) * 8] =
                *(const uint4*)(Ab + (size_t)(u0 >> 2) * K + kb + (u0 & 3) * 8);
            *(uint4*)&sm.A[0][(u1 >> 2) * BK + (u1 & 3) * 8] =
                *(const uint4*)(Ab + (size_t)(u1 >> 2) * K + kb + (u1 & 3) * 8);
            *(uint4*)&sm.B[0][(t >> 2) * BK + (t & 3) * 8] =
                *(const uint4*)(Bb + (size_t)(t >> 2) * K + kb + (t & 3) * 8);
        }
        __syncthreads();
        {
            const unsigned short* sa = sm.A[0];
            const unsigned short* sb = sm.B[0];
            v16bf a0 = load_frag_a(sa + (wm * 32 + ln) * BK, half);
            v16bf a1 = load_frag_a(sa + (wm * 32 + 16 + ln) * BK, half);
            v16bf b0 = load_frag_b(sb + (wn * 32 + ln) * BK + half * 16);
            v16bf b1 = load_frag_b(sb + (wn * 32 + 16 + ln) * BK + half * 16);
            acc00 = wmma_bf16(a0, b0, acc00);
            acc01 = wmma_bf16(a0, b1, acc01);
            acc10 = wmma_bf16(a1, b0, acc10);
            acc11 = wmma_bf16(a1, b1, acc11);
        }
        __syncthreads();
    }
#endif

    // epilogue
#pragma unroll
    for (int i = 0; i < 2; ++i) {
#pragma unroll
        for (int j = 0; j < 2; ++j) {
            const v8f acc = (i == 0) ? (j == 0 ? acc00 : acc01)
                                     : (j == 0 ? acc10 : acc11);
            const int col = n0 + wn * 32 + j * 16 + ln;
            const float bv = bias[col];
#pragma unroll
            for (int v = 0; v < 8; ++v) {
                const int row = m0 + wm * 32 + i * 16 + half * 8 + v;
                const float val = (acc[v] + bv) * alpha;
                if constexpr (MODE == 0) {
                    ((float*)Cout)[(size_t)row * N + col] = val;
                } else if constexpr (MODE == 1) {
                    ((unsigned short*)Cout)[(size_t)row * N + col] = f2bf(val);
                } else {
                    const int b = row >> 10, s = row & 1023;
                    const int h = col >> 6,  d = col & 63;
                    ((unsigned short*)Cout)
                        [(size_t)((b * 16 + h) * 64 + d) * 1024 + s] = f2bf(val);
                }
            }
        }
    }
}

// ---------------------------------------------------------------------------
// Attention core: one (b,h) head and one 16-row query block per workgroup.
//  phase 1: scores = Q K^T (+ rel bias, + mask) -> f32 LDS [16][1024]
//  phase 2: row softmax (register-staged, wave32 shfl reductions) -> bf16 LDS
//  phase 3: ctx = probs @ V (V pre-transposed per head)
// ---------------------------------------------------------------------------
__global__ __launch_bounds__(256) void attn_core_kernel(
    const unsigned short* __restrict__ Q,    // [4096,1024] bf16 (scale folded)
    const unsigned short* __restrict__ Kb,   // [4096,1024] bf16
    const unsigned short* __restrict__ Vt,   // [B,H,64,1024] bf16
    const int*            __restrict__ mask, // [B,1024]
    const float*          __restrict__ rel,  // [127,16]
    unsigned short*       __restrict__ ctx)  // [4096,1024] bf16
{
    constexpr int S = 1024, H = 16;
    __shared__ float sc[16][S];              // 64 KB, reused for bf16 probs

    const int bh = blockIdx.x;
    const int b  = bh >> 4;
    const int h  = bh & 15;
    const int q0 = blockIdx.y * 16;

    const int lane = threadIdx.x & 31;
    const int wave = threadIdx.x >> 5;
    const int half = lane >> 4;
    const int ln   = lane & 15;

    // ---- phase 1: score tiles -------------------------------------------
    const unsigned short* qrow = Q + (size_t)(b * S + q0 + ln) * 1024 + h * 64;
    for (int t = wave; t < S / 16; t += 8) {
        const unsigned short* krow =
            Kb + (size_t)(b * S + t * 16 + ln) * 1024 + h * 64 + half * 16;
        v8f acc = {};
#pragma unroll
        for (int kk = 0; kk < 64; kk += 32) {
            v16bf af = load_frag_a(qrow + kk, half);
            v16bf bf = load_frag_b(krow + kk);
            acc = wmma_bf16(af, bf, acc);
        }
        const int kcol = t * 16 + ln;
        const float mpen = (mask[b * S + kcol] == 0) ? -1e9f : 0.0f;
#pragma unroll
        for (int v = 0; v < 8; ++v) {
            const int qr = half * 8 + v;
            int r = kcol - (q0 + qr) + 63;
            r = r < 0 ? 0 : (r > 126 ? 126 : r);
            sc[qr][kcol] = acc[v] + rel[r * 16 + h] + mpen;
        }
    }
    __syncthreads();

    // ---- phase 2: softmax (2 rows per wave) ------------------------------
    float rv[2][32];
#pragma unroll
    for (int r = 0; r < 2; ++r) {
        const int row = wave * 2 + r;
        float mx = -1e30f;
#pragma unroll
        for (int i = 0; i < 32; ++i) {
            rv[r][i] = sc[row][lane + 32 * i];
            mx = fmaxf(mx, rv[r][i]);
        }
        for (int off = 16; off > 0; off >>= 1)
            mx = fmaxf(mx, __shfl_xor(mx, off, 32));
        float sum = 0.f;
#pragma unroll
        for (int i = 0; i < 32; ++i) {
            rv[r][i] = __expf(rv[r][i] - mx);
            sum += rv[r][i];
        }
        for (int off = 16; off > 0; off >>= 1)
            sum += __shfl_xor(sum, off, 32);
        const float inv = 1.0f / sum;
#pragma unroll
        for (int i = 0; i < 32; ++i) rv[r][i] *= inv;
    }
    __syncthreads();                          // all score reads done
    unsigned short* pr = (unsigned short*)sc; // bf16 probs [16][1024]
#pragma unroll
    for (int r = 0; r < 2; ++r) {
        const int row = wave * 2 + r;
#pragma unroll
        for (int i = 0; i < 32; ++i)
            pr[row * S + lane + 32 * i] = f2bf(rv[r][i]);
    }
    __syncthreads();

    // ---- phase 3: ctx = probs @ V  (waves 0..3, one 16x16 d-tile each) ---
    if (wave < 4) {
        const int d0 = wave * 16;
        const unsigned short* vrow =
            Vt + ((size_t)(b * H + h) * 64 + d0 + ln) * 1024 + half * 16;
        const unsigned short* prow = pr + (size_t)ln * S;
        v8f acc = {};
        for (int kb = 0; kb < S; kb += 32) {
            v16bf af = load_frag_a(prow + kb, half);
            v16bf bf = load_frag_b(vrow + kb);
            acc = wmma_bf16(af, bf, acc);
        }
#pragma unroll
        for (int v = 0; v < 8; ++v) {
            const int qr = half * 8 + v;
            ctx[(size_t)(b * S + q0 + qr) * 1024 + h * 64 + d0 + ln] =
                f2bf(acc[v]);
        }
    }
}

// ---------------------------------------------------------------------------
// launch
// ---------------------------------------------------------------------------
extern "C" void kernel_launch(void* const* d_in, const int* in_sizes, int n_in,
                              void* d_out, int out_size, void* d_ws, size_t ws_size,
                              hipStream_t stream) {
    const float* x    = (const float*)d_in[0];
    const int*   mask = (const int*)  d_in[1];
    const float* wq   = (const float*)d_in[2];
    const float* bq   = (const float*)d_in[3];
    const float* wk   = (const float*)d_in[4];
    const float* bk   = (const float*)d_in[5];
    const float* wv   = (const float*)d_in[6];
    const float* bv   = (const float*)d_in[7];
    const float* wo   = (const float*)d_in[8];
    const float* bo   = (const float*)d_in[9];
    const float* rel  = (const float*)d_in[10];

    char* ws = (char*)d_ws;
    const size_t MB = 1u << 20;
    unsigned short* xbf   = (unsigned short*)(ws + 0 * MB);   // 8 MB
    unsigned short* wqbf  = (unsigned short*)(ws + 8 * MB);   // 2 MB
    unsigned short* wkbf  = (unsigned short*)(ws + 10 * MB);  // 2 MB
    unsigned short* wvbf  = (unsigned short*)(ws + 12 * MB);  // 2 MB
    unsigned short* wobf  = (unsigned short*)(ws + 14 * MB);  // 2 MB
    unsigned short* Qbf   = (unsigned short*)(ws + 16 * MB);  // 8 MB
    unsigned short* Kbf   = (unsigned short*)(ws + 24 * MB);  // 8 MB
    unsigned short* Vtbf  = (unsigned short*)(ws + 32 * MB);  // 8 MB
    unsigned short* ctxbf = (unsigned short*)(ws + 40 * MB);  // 8 MB

    const int NX = 4 * 1024 * 1024;   // x elements
    const int NW = 1024 * 1024;       // weight elements
    dim3 blk(256);

    cvt_bf16_kernel<<<(NX + 255) / 256, blk, 0, stream>>>(x,  xbf,  NX);
    cvt_bf16_kernel<<<(NW + 255) / 256, blk, 0, stream>>>(wq, wqbf, NW);
    cvt_bf16_kernel<<<(NW + 255) / 256, blk, 0, stream>>>(wk, wkbf, NW);
    cvt_bf16_kernel<<<(NW + 255) / 256, blk, 0, stream>>>(wv, wvbf, NW);
    cvt_bf16_kernel<<<(NW + 255) / 256, blk, 0, stream>>>(wo, wobf, NW);

    // block tile 128x64 -> grid (4096/128, 1024/64) = (32, 16)
    dim3 ggrid(32, 16);
    const float inv_sqrt_dk = 0.125f;   // 1/sqrt(64)
    gemm_bf16_tn_kernel<1><<<ggrid, blk, 0, stream>>>(xbf, wqbf, bq, Qbf, inv_sqrt_dk);
    gemm_bf16_tn_kernel<1><<<ggrid, blk, 0, stream>>>(xbf, wkbf, bk, Kbf, 1.0f);
    gemm_bf16_tn_kernel<2><<<ggrid, blk, 0, stream>>>(xbf, wvbf, bv, Vtbf, 1.0f);

    attn_core_kernel<<<dim3(64, 64), blk, 0, stream>>>(Qbf, Kbf, Vtbf, mask, rel, ctxbf);

    gemm_bf16_tn_kernel<0><<<ggrid, blk, 0, stream>>>(ctxbf, wobf, bo, d_out, 1.0f);
}
